// NewCustomDense_8770323218908
// MI455X (gfx1250) — 8-bit matrix operand order repaired and checked by reading, compiled, not run on hardware
//
#include <hip/hip_runtime.h>


typedef __attribute__((ext_vector_type(8))) int v8i;

#if defined(__has_builtin)
#if __has_builtin(__builtin_amdgcn_global_load_async_to_lds_b128) && \
    __has_builtin(__builtin_amdgcn_s_wait_asynccnt)
#define HAVE_ASYNC_LDS 1
#endif
#endif

#define M_B     64
#define K_ORIG  1024
#define N_UNITS 1024
#define KEXP    (K_ORIG * 16)
#define KBLK    64
#define NKB     (KEXP / KBLK)
#define MTILES  (M_B / 16)
#define NTILES  (N_UNITS / 16)
#define APACK_BYTES ((size_t)MTILES * NKB * 1024)
#define QPACK_BYTES ((size_t)NKB * NTILES * 1024)

__global__ void prep_a_kernel(const float* __restrict__ x,
                              unsigned int* __restrict__ Apack) {
  int idx  = blockIdx.x * blockDim.x + threadIdx.x;
  int vg   = idx & 7;
  int lane = (idx >> 3) & 31;
  int kb   = (idx >> 8) & (NKB - 1);
  int mt   = idx >> 16;
  int M    = lane & 15;
  int hi   = lane >> 4;
  int K64  = (vg >> 1) * 16 + hi * 8 + (vg & 1) * 4;
  int kk   = kb * KBLK + K64;
  int k    = kk >> 4;
  int vb   = kk & 15;
  int xv   = (int)x[(mt * 16 + M) * K_ORIG + k];
  unsigned int dw = 0;
  int diff = xv - vb;
  if (diff >= 0 && diff < 4) dw = 1u << (8 * diff);
  Apack[idx] = dw;
}

__global__ void prep_b_kernel(const float* __restrict__ w,
                              unsigned int* __restrict__ Qpack) {
  int idx  = blockIdx.x * blockDim.x + threadIdx.x;
  int vg   = idx & 7;
  int lane = (idx >> 3) & 31;
  int nt   = (idx >> 8) & (NTILES - 1);
  int kb   = idx >> 14;
  int N    = nt * 16 + (lane & 15);
  int hi   = lane >> 4;
  int K64  = (vg >> 1) * 16 + hi * 8 + (vg & 1) * 4;
  int kk   = kb * KBLK + K64;
  int k    = kk >> 4;
  int vb   = kk & 15;
  int wv   = (int)w[k * N_UNITS + N];
  int s    = (wv > 0) - (wv < 0);
  int Wm   = wv < 0 ? -wv : wv;
  unsigned int dw = 0;
#pragma unroll
  for (int i = 0; i < 4; ++i) {
    int q = s * (((vb + i) * Wm) >> 8);
    dw |= ((unsigned int)(q & 0xFF)) << (8 * i);
  }
  Qpack[idx] = dw;
}

__global__ void __launch_bounds__(256)
gemm_iu8_wmma(const unsigned char* __restrict__ Apack,
              const unsigned char* __restrict__ Qpack,
              const float* __restrict__ bias,
              float* __restrict__ out) {
  int tid  = threadIdx.x;
  int lane = tid & 31;
  int wib  = tid >> 5;
  int wave = (blockIdx.x * 256 + tid) >> 5;
  int mt   = wave >> 6;
  int nt   = wave & (NTILES - 1);

  const unsigned char* ap = Apack + ((size_t)mt * NKB) * 1024 + (size_t)lane * 32;
  const unsigned char* bp = Qpack + ((size_t)nt) * 1024 + (size_t)lane * 32;
  const size_t bstride = (size_t)NTILES * 1024;

  int   N  = nt * 16 + (lane & 15);
  float bv = bias[N];

  v8i acc = {};

#ifdef HAVE_ASYNC_LDS
  __shared__ unsigned char Abuf[8][4][1024];
  typedef int v4i __attribute__((vector_size(16)));
  typedef __attribute__((address_space(1))) v4i* gv4_t;
  typedef __attribute__((address_space(3))) v4i* lv4_t;

  lv4_t lbuf0 = (lv4_t)(&Abuf[wib][0][lane * 32]);
  lv4_t lbuf1 = (lv4_t)(&Abuf[wib][1][lane * 32]);
  lv4_t lbuf2 = (lv4_t)(&Abuf[wib][2][lane * 32]);
  lv4_t lbuf3 = (lv4_t)(&Abuf[wib][3][lane * 32]);
  lv4_t lbuf[4] = {lbuf0, lbuf1, lbuf2, lbuf3};

  gv4_t g = (gv4_t)(ap);
#pragma unroll
  for (int j = 0; j < 3; ++j) {
    __builtin_amdgcn_global_load_async_to_lds_b128(g,     lbuf[j],     0, 0);
    __builtin_amdgcn_global_load_async_to_lds_b128(g + 1, lbuf[j] + 1, 0, 0);
    g += 64;
  }
  v8i bcur  = *(const v8i*)bp; bp += bstride;
  v8i bnext = *(const v8i*)bp; bp += bstride;
  __builtin_amdgcn_s_wait_asynccnt(4);
  v8i a = *(const v8i*)&Abuf[wib][0][lane * 32];

#pragma unroll 4
  for (int kb = 0; kb < NKB - 4; ++kb) {
    lv4_t l = lbuf[(kb + 3) & 3];
    __builtin_amdgcn_global_load_async_to_lds_b128(g,     l,     0, 0);
    __builtin_amdgcn_global_load_async_to_lds_b128(g + 1, l + 1, 0, 0);
    g += 64;
    __builtin_amdgcn_s_wait_asynccnt(4);
    v8i anext = *(const v8i*)&Abuf[wib][(kb + 1) & 3][lane * 32];
    __builtin_prefetch(bp + 6 * bstride, 0, 0);
    acc = __builtin_amdgcn_wmma_i32_16x16x64_iu8(false, a, true, bcur, acc, false, false);
    a = anext;
    bcur = bnext;
    bnext = *(const v8i*)bp; bp += bstride;
  }
  {
    __builtin_amdgcn_global_load_async_to_lds_b128(g,     lbuf3,     0, 0);
    __builtin_amdgcn_global_load_async_to_lds_b128(g + 1, lbuf3 + 1, 0, 0);
    __builtin_amdgcn_s_wait_asynccnt(4);
    v8i anext = *(const v8i*)&Abuf[wib][1][lane * 32];
    acc = __builtin_amdgcn_wmma_i32_16x16x64_iu8(false, a, true, bcur, acc, false, false);
    a = anext; bcur = bnext;
    bnext = *(const v8i*)bp; bp += bstride;
  }
  {
    __builtin_amdgcn_s_wait_asynccnt(2);
    v8i anext = *(const v8i*)&Abuf[wib][2][lane * 32];
    acc = __builtin_amdgcn_wmma_i32_16x16x64_iu8(false, a, true, bcur, acc, false, false);
    a = anext; bcur = bnext;
    bnext = *(const v8i*)bp;
  }
  {
    __builtin_amdgcn_s_wait_asynccnt(0);
    v8i anext = *(const v8i*)&Abuf[wib][3][lane * 32];
    acc = __builtin_amdgcn_wmma_i32_16x16x64_iu8(false, a, true, bcur, acc, false, false);
    a = anext; bcur = bnext;
  }
  acc = __builtin_amdgcn_wmma_i32_16x16x64_iu8(false, a, true, bcur, acc, false, false);
#else
#pragma unroll 4
  for (int kb = 0; kb < NKB; ++kb) {
    v8i a = *(const v8i*)ap;
    v8i b = *(const v8i*)bp;
    __builtin_prefetch(bp + 8 * bstride, 0, 0);
    acc = __builtin_amdgcn_wmma_i32_16x16x64_iu8(false, a, true, b, acc, false, false);
    ap += 1024;
    bp += bstride;
  }
#endif

  int Mb = mt * 16 + (lane >> 4) * 8;
#pragma unroll
  for (int r = 0; r < 8; ++r) {
    float v = (float)acc[r] + bv;
    out[(size_t)(Mb + r) * N_UNITS + N] = v > 0.f ? v : 0.f;
  }
}

extern "C" void kernel_launch(void* const* d_in, const int* in_sizes, int n_in,
                              void* d_out, int out_size, void* d_ws, size_t ws_size,
                              hipStream_t stream) {
  (void)in_sizes; (void)n_in; (void)out_size; (void)ws_size;
  const float* x    = (const float*)d_in[0];
  const float* w    = (const float*)d_in[1];
  const float* bias = (const float*)d_in[2];

  unsigned char* Apack = (unsigned char*)d_ws;
  unsigned char* Qpack = Apack + APACK_BYTES;

  int a_dwords = (int)(APACK_BYTES / 4);
  int q_dwords = (int)(QPACK_BYTES / 4);
  prep_a_kernel<<<a_dwords / 256, 256, 0, stream>>>(x, (unsigned int*)Apack);
  prep_b_kernel<<<q_dwords / 256, 256, 0, stream>>>(w, (unsigned int*)Qpack);

  int gemm_threads = MTILES * NTILES * 32;
  gemm_iu8_wmma<<<gemm_threads / 256, 256, 0, stream>>>(Apack, Qpack, bias, (float*)d_out);
}
